// SAMSNLM_58162447122724
// MI455X (gfx1250) — matrix operand order repaired and checked by reading, not compiled, not run on hardware
//
#include <hip/hip_runtime.h>
#include <hip/hip_bf16.h>

typedef _Float16 half_t;
typedef __attribute__((ext_vector_type(16))) _Float16 v16h;
typedef __attribute__((ext_vector_type(8)))  float    v8f;
typedef unsigned int u32x4 __attribute__((ext_vector_type(4)));
typedef int          i32x8 __attribute__((ext_vector_type(8)));
typedef int          i32x4 __attribute__((ext_vector_type(4)));

#define BATCH 2
#define CH    256
#define HS    96
#define HW    (HS * HS)     // 9216
#define NTOK  HW
#define INNER 14
#define NDIL  3
#define KT    32            // keys per attention tile

// ---------------------------------------------------------------- WMMA helpers

static __device__ inline v8f wmma16(v16h a, v16h b, v8f c) {
  // D = A(16x32 f16) * B(32x16 f16) + C(16x16 f32)
  return __builtin_amdgcn_wmma_f32_16x16x32_f16(
      /*neg_a=*/false, a, /*neg_b=*/false, b,
      /*c_mod=*/(short)0, c, /*reuse_a=*/false, /*reuse_b=*/false);
}

// A fragment (16x32, MxK) from a row-major matrix, leading dim ld.
// Per lane: two contiguous 16B chunks -> global_load_b128 / ds_load_b128.
static __device__ inline v16h load_A_rm(const half_t* base, int row0, int k0, int ld) {
  const int lane = threadIdx.x & 31;
  const int m    = lane & 15;
  const int kh   = (lane >> 4) << 4;     // 0 or 16
  const half_t* p = base + (long)(row0 + m) * ld + k0;
  v16h a;
#pragma unroll
  for (int i = 0; i < 8; ++i) a[i] = p[kh + i];
#pragma unroll
  for (int i = 0; i < 8; ++i) a[8 + i] = p[kh + 8 + i];
  return a;
}

// B fragment (32x16, KxN) where B^T is stored row-major (element(k,n)=base[(n0+n)*ld+k]).
// Per lane: one contiguous 32B chunk -> 2x b128 loads.
static __device__ inline v16h load_Bt_rm(const half_t* base, int n0, int k0, int ld) {
  const int lane = threadIdx.x & 31;
  const int nn   = lane & 15;
  const int kb   = (lane >> 4) << 4;
  const half_t* p = base + (long)(n0 + nn) * ld + k0 + kb;
  v16h b;
#pragma unroll
  for (int i = 0; i < 16; ++i) b[i] = p[i];
  return b;
}

// B fragment (32x16, KxN) from a K-major matrix: element(k,n)=base[k*ldb+n].
// (Only used against LDS in the attention P*V step.)
static __device__ inline v16h load_B_km(const half_t* base, int k0, int n0, int ldb) {
  const int lane = threadIdx.x & 31;
  const int nn   = lane & 15;
  const int kb   = (lane >> 4) << 4;     // 0 or 16
  v16h b;
#pragma unroll
  for (int i = 0; i < 16; ++i) b[i] = base[(long)(k0 + kb + i) * ldb + n0 + nn];
  return b;
}

// ------------------------------------------------------------ TDM tile loader
// Async DMA of a KT x CH f16 tile (row-major, ld = CH elements) into LDS.
// D# per CDNA5 ISA ch.8. clang-23 toolchain: 6-arg builtin.

static __device__ inline void tdm_load_tile(const half_t* gsrc, unsigned int lds_off) {
  unsigned long long ga = (unsigned long long)(uintptr_t)gsrc;
  u32x4 g0;
  g0[0] = 1u;                                    // count=1, no gather/iterate
  g0[1] = lds_off;                               // LDS byte address
  g0[2] = (unsigned int)ga;                      // global_addr[31:0]
  g0[3] = (unsigned int)((ga >> 32) & 0x01FFFFFFu) | (2u << 30);  // [56:32] | type=2
  i32x8 g1;
  g1[0] = (1 << 16);                             // data_size=1 (2 bytes), mask=0
  g1[1] = (CH & 0xFFFF) << 16;                   // tensor_dim0[15:0] @ bits 63:48
  g1[2] = (NTOK & 0xFFFF) << 16;                 // tensor_dim1[15:0] @ bits 95:80
  g1[3] = (CH & 0xFFFF) << 16;                   // tile_dim0 @ bits 127:112
  g1[4] = KT;                                    // tile_dim1 @ bits 143:128
  g1[5] = CH;                                    // tensor_dim0_stride[31:0]
  g1[6] = 0;
  g1[7] = 0;
  i32x4 z4 = {0, 0, 0, 0};
  i32x8 z8 = {0, 0, 0, 0, 0, 0, 0, 0};
  __builtin_amdgcn_tensor_load_to_lds(g0, g1, z4, z4, z8, 0);
}

// ------------------------------------------------ weight convert f32->f16 pad

__global__ void k_cvt_pad(const float* __restrict__ src, half_t* __restrict__ dst,
                          int srows, int scols, int drows, int dcols) {
  long i = (long)blockIdx.x * blockDim.x + threadIdx.x;
  long total = (long)drows * dcols;
  if (i >= total) return;
  int r = (int)(i / dcols), c = (int)(i % dcols);
  float v = (r < srows && c < scols) ? src[(long)r * scols + c] : 0.0f;
  dst[i] = (half_t)v;
}

// ------------------------------- x: NCHW f32 -> pixel-major (B,HW,C) f16

__global__ void k_cvt_x(const float* __restrict__ x, half_t* __restrict__ xt) {
  long i = (long)blockIdx.x * blockDim.x + threadIdx.x;
  const long total = (long)BATCH * HW * CH;
  if (i >= total) return;
  int c   = (int)(i % CH);
  int pix = (int)((i / CH) % HW);
  int b   = (int)(i / ((long)CH * HW));
  xt[i] = (half_t)x[((long)b * CH + c) * HW + pix];   // write coalesced
}

// ---------------------- proj 1x1: D(pix x oc) = X(pix x 256) * W^T, + bias

__global__ __launch_bounds__(256) void k_proj(const half_t* __restrict__ xt,
                                              const half_t* __restrict__ wh,
                                              const float* __restrict__ bias,
                                              float* __restrict__ fmap,    // (B,HW,C) f32
                                              half_t* __restrict__ fmaph) { // (B,HW,C) f16
  int wid = (int)(((long)blockIdx.x * blockDim.x + threadIdx.x) >> 5);
  const int ptiles = HW / 16;
  int pt = wid % ptiles;
  int t  = wid / ptiles;
  int ot = t % (CH / 16);
  int b  = t / (CH / 16);
  if (b >= BATCH) return;
  const half_t* xb = xt + (long)b * HW * CH;
  v8f acc = {};
  for (int k0 = 0; k0 < CH; k0 += 32) {
    __builtin_prefetch(xb + (long)(pt * 16) * CH + k0 + 256, 0, 0);
    v16h a  = load_A_rm(xb, pt * 16, k0, CH);      // contiguous per lane
    v16h bm = load_Bt_rm(wh, ot * 16, k0, CH);     // contiguous per lane
    acc = wmma16(a, bm, acc);
  }
  const int lane = threadIdx.x & 31;
  const int nn = lane & 15, mo = (lane >> 4) << 3;
  const int oc = ot * 16 + nn;
  const float bv = bias[oc];
#pragma unroll
  for (int r = 0; r < 8; ++r) {
    int pix = pt * 16 + mo + r;
    float v = acc[r] + bv;
    long idx = ((long)b * HW + pix) * CH + oc;     // lane-contiguous stores
    fmap[idx]  = v;
    fmaph[idx] = (half_t)v;
  }
}

// ------------- reduce 1x1: D(pix x 16) = fmap(pix x 256) * Wr^T (rows padded)

__global__ __launch_bounds__(256) void k_reduce(const half_t* __restrict__ fmaph,
                                                const half_t* __restrict__ wh,
                                                const float* __restrict__ bias,
                                                float* __restrict__ xrt) {  // (B,HW,16) f32
  int wid = (int)(((long)blockIdx.x * blockDim.x + threadIdx.x) >> 5);
  const int ptiles = HW / 16;
  int pt = wid % ptiles;
  int b  = wid / ptiles;
  if (b >= BATCH) return;
  const half_t* fb = fmaph + (long)b * HW * CH;
  v8f acc = {};
  for (int k0 = 0; k0 < CH; k0 += 32) {
    v16h a  = load_A_rm(fb, pt * 16, k0, CH);
    v16h bm = load_Bt_rm(wh, 0, k0, CH);
    acc = wmma16(a, bm, acc);
  }
  const int lane = threadIdx.x & 31;
  const int nn = lane & 15, mo = (lane >> 4) << 3;
  const float bv = (nn < INNER) ? bias[nn] : 0.0f;
#pragma unroll
  for (int r = 0; r < 8; ++r) {
    int pix = pt * 16 + mo + r;
    float v = (nn < INNER) ? (acc[r] + bv) : 0.0f;
    xrt[((long)b * HW + pix) * 16 + nn] = v;       // lane-contiguous stores
  }
}

// ---------------------------------------- dilated 3x3 convs (14ch, VALU f32)
// xr is pixel-major (B,HW,16) so the cin loop reads contiguously.

__global__ void k_dil(const float* __restrict__ xrt, const float* __restrict__ dw,
                      const float* __restrict__ db, float* __restrict__ co) {
  long i = (long)blockIdx.x * blockDim.x + threadIdx.x;
  const long total = (long)BATCH * NDIL * INNER * HW;
  if (i >= total) return;
  int pix = (int)(i % HW);
  int c   = (int)((i / HW) % INNER);
  int di  = (int)((i / ((long)HW * INNER)) % NDIL);
  int b   = (int)(i / ((long)HW * INNER * NDIL));
  int h = pix / HS, w = pix % HS;
  int d = di + 1;
  float acc = db[di * INNER + c];
  const float* xb = xrt + (long)b * HW * 16;
  const float* wp = dw + (((long)di * INNER + c) * INNER) * 9;
#pragma unroll
  for (int kh = 0; kh < 3; ++kh) {
    int ih = h + (kh - 1) * d;
    if (ih < 0 || ih >= HS) continue;
#pragma unroll
    for (int kw = 0; kw < 3; ++kw) {
      int iw = w + (kw - 1) * d;
      if (iw < 0 || iw >= HS) continue;
      const float* xp = xb + ((long)ih * HS + iw) * 16;
      for (int cin = 0; cin < INNER; ++cin)
        acc += wp[cin * 9 + kh * 3 + kw] * xp[cin];
    }
  }
  co[i] = acc;
}

// ------ feats: co[di] * transform_j(xr) -> pixel-major (B,HW,256) f16, padded

__global__ void k_feats(const float* __restrict__ xrt, const float* __restrict__ co,
                        half_t* __restrict__ feats) {
  long i = (long)blockIdx.x * blockDim.x + threadIdx.x;
  const long total = (long)BATCH * HW * CH;
  if (i >= total) return;
  int q   = (int)(i % CH);
  int pix = (int)((i / CH) % HW);
  int b   = (int)(i / ((long)CH * HW));
  half_t v = (half_t)0.0f;
  if (q < INNER * 6 * NDIL) {                       // 252 real channels
    int c  = q % INNER;
    int tj = (q / INNER) % 6;
    int di = q / (INNER * 6);
    int h = pix / HS, w = pix % HS;
    int sh, sw;
    switch (tj) {
      default:
      case 0: sh = h;          sw = w;          break;  // identity
      case 1: sh = h;          sw = HS - 1 - w; break;  // flip -1
      case 2: sh = HS - 1 - h; sw = w;          break;  // flip -2
      case 3: sh = w;          sw = HS - 1 - h; break;  // rot90 k=1
      case 4: sh = HS - 1 - h; sw = HS - 1 - w; break;  // rot90 k=2
      case 5: sh = HS - 1 - w; sw = h;          break;  // rot90 k=3
    }
    float t  = xrt[((long)b * HW + sh * HS + sw) * 16 + c];
    float cv = co[(((long)b * NDIL + di) * INNER + c) * HW + pix];
    v = (half_t)(cv * t);
  }
  feats[i] = v;                                     // write coalesced
}

// ------------- fuse 1x1: D(pix x oc) = feats(pix x 256) * Wf^T -> tok (pix,C)

__global__ __launch_bounds__(256) void k_fuse(const half_t* __restrict__ feats,
                                              const half_t* __restrict__ wh,
                                              const float* __restrict__ bias,
                                              half_t* __restrict__ tok) {
  int wid = (int)(((long)blockIdx.x * blockDim.x + threadIdx.x) >> 5);
  const int ptiles = HW / 16;
  int pt = wid % ptiles;
  int t  = wid / ptiles;
  int ot = t % (CH / 16);
  int b  = t / (CH / 16);
  if (b >= BATCH) return;
  const half_t* fb = feats + (long)b * HW * CH;
  v8f acc = {};
  for (int k0 = 0; k0 < CH; k0 += 32) {
    v16h a  = load_A_rm(fb, pt * 16, k0, CH);
    v16h bm = load_Bt_rm(wh, ot * 16, k0, CH);
    acc = wmma16(a, bm, acc);
  }
  const int lane = threadIdx.x & 31;
  const int nn = lane & 15, mo = (lane >> 4) << 3;
  const int oc = ot * 16 + nn;
  const float bv = bias[oc];
#pragma unroll
  for (int r = 0; r < 8; ++r) {
    int pix = pt * 16 + mo + r;
    tok[((long)b * NTOK + pix) * CH + oc] = (half_t)(acc[r] + bv);
  }
}

// ------------------------- flash self-attention + residual (dominant kernel)
// One wave = 16 query rows; 8 waves/WG share each 32x256 K/V tile, which is
// DMA'd into LDS by the Tensor Data Mover (double-buffered, s_wait_tensorcnt).

__global__ __launch_bounds__(256, 1) void k_attn(const half_t* __restrict__ tok,
                                                 const float* __restrict__ fmap,  // (B,HW,C)
                                                 float* __restrict__ outp) {      // NCHW
  __shared__ _Float16 vbuf[2][KT][CH];   // 2 x 16KB double-buffered K/V tile
  __shared__ _Float16 pbuf[8][16][34];   // per-wave P-transpose staging
  const int wave = threadIdx.x >> 5;
  const int lane = threadIdx.x & 31;
  const int nn   = lane & 15;
  const int mo   = (lane >> 4) << 3;

  int wid = blockIdx.x * 8 + wave;
  const int qtiles = NTOK / 16;          // 576 per batch; 72 blocks per batch
  int b  = wid / qtiles;
  int q0 = (wid % qtiles) * 16;
  const half_t* tokb = tok + (long)b * NTOK * CH;

  // Q is fixed per wave: hoist all 8 A-fragments (K-dim 256) into registers.
  v16h qf[8];
#pragma unroll
  for (int kk = 0; kk < 8; ++kk) qf[kk] = load_A_rm(tokb, q0, kk * 32, CH);

  float m_i[8], l_i[8];
#pragma unroll
  for (int r = 0; r < 8; ++r) { m_i[r] = -3.0e38f; l_i[r] = 0.0f; }
  v8f zero = {};
  v8f O[16];
#pragma unroll
  for (int cb = 0; cb < 16; ++cb) O[cb] = zero;

  const float scale = 0.0625f;           // 1/sqrt(256)

  const unsigned int vb0 = (unsigned int)(uintptr_t)&vbuf[0][0][0];
  const unsigned int vb1 = (unsigned int)(uintptr_t)&vbuf[1][0][0];
  if (wave == 0) tdm_load_tile(tokb, vb0);            // prime tile 0

  const int NT = NTOK / KT;              // 288 key tiles
  for (int t = 0; t < NT; ++t) {
    if (wave == 0) {
      if (t + 1 < NT) {
        // issue next tile into the other buffer, then wait for current tile
        tdm_load_tile(tokb + (long)(t + 1) * KT * CH, (t & 1) ? vb0 : vb1);
        __builtin_amdgcn_s_wait_tensorcnt(1);
      } else {
        __builtin_amdgcn_s_wait_tensorcnt(0);
      }
    }
    __syncthreads();                     // publish tile t to all 8 waves
    const _Float16* vt = &vbuf[t & 1][0][0];

    // S = Q * K^T for 32 keys (two 16x16 tiles); K fragments from LDS (b128)
    v8f S0 = {}, S1 = {};
#pragma unroll
    for (int kk = 0; kk < 8; ++kk) {
      v16h bk0 = load_Bt_rm(vt, 0,  kk * 32, CH);
      v16h bk1 = load_Bt_rm(vt, 16, kk * 32, CH);
      S0 = wmma16(qf[kk], bk0, S0);
      S1 = wmma16(qf[kk], bk1, S1);
    }

    // online softmax over the 32 new columns (row lives across 16 lanes)
    float alpha[8];
#pragma unroll
    for (int r = 0; r < 8; ++r) {
      float s0 = S0[r] * scale, s1 = S1[r] * scale;
      float v = fmaxf(s0, s1);
      v = fmaxf(v, __shfl_xor(v, 8, 16));
      v = fmaxf(v, __shfl_xor(v, 4, 16));
      v = fmaxf(v, __shfl_xor(v, 2, 16));
      v = fmaxf(v, __shfl_xor(v, 1, 16));
      float mnew = fmaxf(m_i[r], v);
      alpha[r] = __expf(m_i[r] - mnew);
      float p0 = __expf(s0 - mnew);
      float p1 = __expf(s1 - mnew);
      float rs = p0 + p1;
      rs += __shfl_xor(rs, 8, 16);
      rs += __shfl_xor(rs, 4, 16);
      rs += __shfl_xor(rs, 2, 16);
      rs += __shfl_xor(rs, 1, 16);
      l_i[r] = l_i[r] * alpha[r] + rs;
      m_i[r] = mnew;
      pbuf[wave][mo + r][nn]      = (half_t)p0;   // D-layout -> LDS (row-major)
      pbuf[wave][mo + r][16 + nn] = (half_t)p1;
    }

    // rescale running output
#pragma unroll
    for (int cb = 0; cb < 16; ++cb) {
#pragma unroll
      for (int r = 0; r < 8; ++r) O[cb][r] *= alpha[r];
    }

    // O += P(16x32) * V(32x256); P from LDS in A-layout, V fragments from LDS
    v16h pa = load_A_rm(&pbuf[wave][0][0], 0, 0, 34);
#pragma unroll
    for (int cb = 0; cb < 16; ++cb) {
      v16h bv = load_B_km(vt, 0, cb * 16, CH);
      O[cb] = wmma16(pa, bv, O[cb]);
    }
    __syncthreads();                     // all waves done with tile t
  }

  // epilogue: out = fmap + 0.2 * (O / l); fmap read coalesced (pixel-major),
  // output scattered only because the required result layout is NCHW.
#pragma unroll
  for (int cb = 0; cb < 16; ++cb) {
    int c = cb * 16 + nn;
#pragma unroll
    for (int r = 0; r < 8; ++r) {
      int q = q0 + mo + r;
      float f = fmap[((long)b * HW + q) * CH + c];
      outp[((long)b * CH + c) * HW + q] = f + 0.2f * (O[cb][r] / l_i[r]);
    }
  }
}

// ---------------------------------------------------------------- launcher

extern "C" void kernel_launch(void* const* d_in, const int* in_sizes, int n_in,
                              void* d_out, int out_size, void* d_ws, size_t ws_size,
                              hipStream_t stream) {
  (void)in_sizes; (void)n_in; (void)out_size; (void)ws_size;
  const float* x        = (const float*)d_in[0];
  const float* proj_w   = (const float*)d_in[1];
  const float* proj_b   = (const float*)d_in[2];
  const float* reduce_w = (const float*)d_in[3];
  const float* reduce_b = (const float*)d_in[4];
  const float* dil_w    = (const float*)d_in[5];
  const float* dil_b    = (const float*)d_in[6];
  const float* fuse_w   = (const float*)d_in[7];
  const float* fuse_b   = (const float*)d_in[8];
  float* out = (float*)d_out;

  char* ws = (char*)d_ws;
  size_t off = 0;
  auto take = [&](size_t bytes) -> char* {
    char* p = ws + off;
    off = (off + bytes + 255) & ~(size_t)255;
    return p;
  };
  half_t* xt     = (half_t*)take((size_t)BATCH * HW * CH * sizeof(half_t));
  half_t* pwh    = (half_t*)take((size_t)CH * CH * sizeof(half_t));
  half_t* rwh    = (half_t*)take((size_t)16 * CH * sizeof(half_t));
  half_t* fwh    = (half_t*)take((size_t)CH * CH * sizeof(half_t));
  float*  fmap   = (float*) take((size_t)BATCH * HW * CH * sizeof(float));
  half_t* fmaph  = (half_t*)take((size_t)BATCH * HW * CH * sizeof(half_t));
  float*  xrt    = (float*) take((size_t)BATCH * HW * 16 * sizeof(float));
  float*  co     = (float*) take((size_t)BATCH * NDIL * INNER * HW * sizeof(float));
  half_t* featsh = (half_t*)take((size_t)BATCH * HW * CH * sizeof(half_t));
  half_t* tokh   = (half_t*)take((size_t)BATCH * NTOK * CH * sizeof(half_t));

  const int T = 256;
  // input transpose+convert and weight converts (pad reduce rows, fuse cols)
  {
    long n = (long)BATCH * HW * CH;
    k_cvt_x<<<(int)((n + T - 1) / T), T, 0, stream>>>(x, xt);
  }
  k_cvt_pad<<<(CH * CH + T - 1) / T, T, 0, stream>>>(proj_w, pwh, CH, CH, CH, CH);
  k_cvt_pad<<<(16 * CH + T - 1) / T, T, 0, stream>>>(reduce_w, rwh, INNER, CH, 16, CH);
  k_cvt_pad<<<(CH * CH + T - 1) / T, T, 0, stream>>>(fuse_w, fwh, CH, INNER * 6 * NDIL, CH, CH);

  // proj: BATCH * (CH/16) * (HW/16) wave-tiles, 8 waves per block
  k_proj<<<BATCH * (CH / 16) * (HW / 16) / 8, T, 0, stream>>>(xt, pwh, proj_b, fmap, fmaph);
  // reduce: BATCH * (HW/16) wave-tiles
  k_reduce<<<BATCH * (HW / 16) / 8, T, 0, stream>>>(fmaph, rwh, reduce_b, xrt);
  // dilated convs
  {
    long n = (long)BATCH * NDIL * INNER * HW;
    k_dil<<<(int)((n + T - 1) / T), T, 0, stream>>>(xrt, dil_w, dil_b, co);
  }
  // feats
  {
    long n = (long)BATCH * HW * CH;
    k_feats<<<(int)((n + T - 1) / T), T, 0, stream>>>(xrt, co, featsh);
  }
  // fuse
  k_fuse<<<BATCH * (CH / 16) * (HW / 16) / 8, T, 0, stream>>>(featsh, fwh, fuse_b, tokh);
  // flash attention + residual (TDM-fed)
  k_attn<<<BATCH * (NTOK / 16) / 8, T, 0, stream>>>(tokh, fmap, out);
}
